// ATOCActorNet_61022895342091
// MI455X (gfx1250) — matrix operand order repaired and checked by reading, not compiled, not run on hardware
//
#include <hip/hip_runtime.h>
#include <hip/hip_bf16.h>
#include <math.h>

// ---------------- dims ----------------
#define B_    64
#define A_    128
#define OBS_  1024
#define TD_   256
#define ACTD_ 64
#define MG_   8
#define EATT_ 128
#define H_    128
#define ROWS_ (B_*A_)        // 8192
#define GATES_ (4*H_)        // 512
#define KC_   64             // K chunk staged in LDS per iteration
#define BUFB_ 10240          // bytes per LDS buffer: A tile 2048 + W tile 8192

typedef __bf16 bf16_t;
typedef __attribute__((ext_vector_type(16))) bf16_t v16bf;
typedef __attribute__((ext_vector_type(8)))  bf16_t v8bf;
typedef __attribute__((ext_vector_type(8)))  float  v8f;

// ---------------- WMMA fragment loaders (wave32, 16x16x32 bf16) ----------------
// A (16x32, MxK): lane L holds row M=L&15; lanes<16: K = k0..k0+15,
// lanes>=16: K = k0+16..k0+31. Two contiguous 16B chunks per lane.
__device__ __forceinline__ v16bf load_a_frag(const bf16_t* __restrict__ row, int k0, int lane) {
  int kb = k0 + ((lane >> 4) << 4);
  v8bf lo = *(const v8bf*)(row + kb);
  v8bf hi = *(const v8bf*)(row + kb + 8);
  return __builtin_shufflevector(lo, hi, 0,1,2,3,4,5,6,7,8,9,10,11,12,13,14,15);
}
// B (32x16, KxN): lane L holds column N=L&15; lanes<16: K=k0..k0+15, lanes>=16: K=k0+16..31.
// W stored (N x K) row-major bf16 -> one contiguous 32B load per lane.
__device__ __forceinline__ v16bf load_b_frag(const bf16_t* __restrict__ row, int k0, int lane) {
  int kb = k0 + ((lane >> 4) << 4);
  return *(const v16bf*)(row + kb);
}

__device__ __forceinline__ v8f wmma32(v16bf a, v16bf b, v8f c) {
  return __builtin_amdgcn_wmma_f32_16x16x32_bf16(false, a, false, b, (short)0, c, false, false);
}

__device__ __forceinline__ float sigm(float x) { return 1.f / (1.f + expf(-x)); }

// async copy 16B global -> LDS (no VGPR data path, tracked by ASYNCcnt)
__device__ __forceinline__ void async_g2l_b128(unsigned lds_off, const void* gaddr) {
  asm volatile("global_load_async_to_lds_b128 %0, %1, off"
               :: "v"(lds_off), "v"((unsigned long long)(uintptr_t)gaddr) : "memory");
}

// ---------------- fp32 -> bf16 conversion (one-time per tensor) ----------------
__global__ void cvt_bf16(const float* __restrict__ x, bf16_t* __restrict__ y, int n) {
  int i = blockIdx.x * blockDim.x + threadIdx.x;
  if (i < n) y[i] = (bf16_t)x[i];
}

// ---------------- cooperative GEMM: C(MxN) = A(MxK) * W(NxK)^T + bias ----------------
// Block = 4 waves, one 16-row M tile, 4 adjacent N tiles. Double-buffered LDS staging:
// chunk i+1 is issued via global_load_async_to_lds_b128 before waiting on chunk i
// (s_wait_asynccnt 5 leaves the 5 in-flight prefetch ops outstanding), so the TDM-style
// async pipe overlaps the WMMA stream.
template<int ACT>  // 0 none, 1 relu
__global__ void gemm_coop(const bf16_t* __restrict__ Abf, int lda,
                          const bf16_t* __restrict__ Wbf, int ldw,
                          const float* __restrict__ bias,
                          float* __restrict__ Cf, bf16_t* __restrict__ Cb, int ldc,
                          int M, int N, int K) {
  extern __shared__ char sm[];               // dynamic LDS only -> base offset 0
  int tid = threadIdx.x, lane = tid & 31, wv = tid >> 5;
  int ngrp = N >> 6;
  int tm  = blockIdx.x / ngrp;
  int tn0 = (blockIdx.x % ngrp) << 2;        // first of 4 N tiles

  auto stage = [&](int kc, int buf) {
    unsigned base = (unsigned)(buf * BUFB_);
    {  // A tile: 16 rows x 64 bf16 = 128 x 16B segments (one per thread)
      int r = tid >> 3, s = tid & 7;
      async_g2l_b128(base + (unsigned)(r * 128 + s * 16),
                     Abf + (size_t)(tm * 16 + r) * lda + kc + s * 8);
    }
#pragma unroll
    for (int it = 0; it < 4; ++it) {         // W tile: 64 rows x 64 bf16 = 512 segs
      int seg = tid + it * 128;
      int r = seg >> 3, s = seg & 7;
      async_g2l_b128(base + 2048u + (unsigned)(r * 128 + s * 16),
                     Wbf + (size_t)(tn0 * 16 + r) * ldw + kc + s * 8);
    }
  };

  v8f acc = {};
  stage(0, 0);
  int cur = 0;
  for (int kc = 0; kc < K; kc += KC_) {
    if (kc + KC_ < K) {
      stage(kc + KC_, cur ^ 1);                               // prefetch next chunk
      asm volatile("s_wait_asynccnt 0x5" ::: "memory");       // wait current, keep prefetch in flight
    } else {
      asm volatile("s_wait_asynccnt 0x0" ::: "memory");
    }
    __syncthreads();
    const bf16_t* aS = (const bf16_t*)(sm + cur * BUFB_);
    const bf16_t* wS = (const bf16_t*)(sm + cur * BUFB_ + 2048);
#pragma unroll
    for (int kk = 0; kk < KC_; kk += 32) {
      v16bf af  = load_a_frag(aS + (size_t)(lane & 15) * KC_, kk, lane);
      v16bf bfr = load_b_frag(wS + (size_t)(wv * 16 + (lane & 15)) * KC_, kk, lane);
      acc = wmma32(af, bfr, acc);
    }
    __syncthreads();
    cur ^= 1;
  }
  int n  = (tn0 + wv) * 16 + (lane & 15);
  int mb = tm * 16 + ((lane >> 4) << 3);
  float bv = bias[n];
#pragma unroll
  for (int r = 0; r < 8; ++r) {
    float v = acc[r] + bv;
    if (ACT == 1) v = fmaxf(v, 0.f);
    if (Cf) Cf[(size_t)(mb + r) * ldc + n] = v;
    if (Cb) Cb[(size_t)(mb + r) * ldc + n] = (bf16_t)v;
  }
}

// ---------------- LayerNorm (one wave per row); optional f32 and/or bf16 outputs ----------------
template<int D, int ACT>  // ACT: 0 none, 1 relu, 2 tanh
__global__ void ln_kernel(const float* __restrict__ X, const float* __restrict__ g,
                          const float* __restrict__ be,
                          float* __restrict__ Yf, bf16_t* __restrict__ Yb, int rows) {
  int wid = (int)((blockIdx.x * blockDim.x + threadIdx.x) >> 5);
  if (wid >= rows) return;
  int lane = threadIdx.x & 31;
  const float* x = X + (size_t)wid * D;
  constexpr int P = D / 32;
  float v[P];
  float s = 0.f;
#pragma unroll
  for (int i = 0; i < P; ++i) { v[i] = x[lane + 32 * i]; s += v[i]; }
#pragma unroll
  for (int o = 16; o > 0; o >>= 1) s += __shfl_xor(s, o, 32);
  float mu = s / (float)D;
  float q = 0.f;
#pragma unroll
  for (int i = 0; i < P; ++i) { float d = v[i] - mu; q += d * d; }
#pragma unroll
  for (int o = 16; o > 0; o >>= 1) q += __shfl_xor(q, o, 32);
  float rs = rsqrtf(q / (float)D + 1e-5f);
#pragma unroll
  for (int i = 0; i < P; ++i) {
    int c = lane + 32 * i;
    float val = (v[i] - mu) * rs * g[c] + be[c];
    if (ACT == 1) val = fmaxf(val, 0.f);
    if (ACT == 2) val = tanhf(val);
    if (Yf) Yf[(size_t)wid * D + c] = val;
    if (Yb) Yb[(size_t)wid * D + c] = (bf16_t)val;
  }
}

// ---------------- attention head: prob = sigmoid(x . w3 + b3) ----------------
__global__ void attn_final(const float* __restrict__ X, const float* __restrict__ w3,
                           const float* __restrict__ b3, float* __restrict__ prob,
                           float* __restrict__ isinit) {
  int r = blockIdx.x * blockDim.x + threadIdx.x;
  if (r >= ROWS_) return;
  float s = 0.f;
  for (int i = 0; i < EATT_; ++i) s += X[(size_t)r * EATT_ + i] * w3[i];
  s += b3[0];
  float p = sigm(s);
  prob[r]   = p;
  isinit[r] = (p > 0.4f) ? 1.f : 0.f;
}

// ---------------- row squared norms of ct ----------------
__global__ void rownorm_kernel(const float* __restrict__ X, float* __restrict__ SQ) {
  int r = blockIdx.x * blockDim.x + threadIdx.x;
  if (r >= ROWS_) return;
  float s = 0.f;
  for (int i = 0; i < TD_; ++i) { float v = X[(size_t)r * TD_ + i]; s += v * v; }
  SQ[r] = s;
}

// ---------------- per-batch Gram + distance (bf16 WMMA) ----------------
__global__ void dist_gemm(const bf16_t* __restrict__ ctb, const float* __restrict__ sq,
                          float* __restrict__ dists) {
  int bidx = blockIdx.y;
  const bf16_t* base = ctb + (size_t)bidx * A_ * TD_;
  int lane   = threadIdx.x & 31;
  int waveId = blockIdx.x * 4 + (threadIdx.x >> 5);   // 64 tiles (8x8)
  int tm = waveId / 8, tn = waveId % 8;
  const bf16_t* pa = base + (size_t)(tm * 16 + (lane & 15)) * TD_;
  const bf16_t* pw = base + (size_t)(tn * 16 + (lane & 15)) * TD_;
  v8f acc = {};
  for (int k = 0; k < TD_; k += 32) {
    acc = wmma32(load_a_frag(pa, k, lane), load_b_frag(pw, k, lane), acc);
  }
  int n  = tn * 16 + (lane & 15);
  int mb = tm * 16 + ((lane >> 4) << 3);
  const float* sqb = sq + bidx * A_;
  float* drow = dists + (size_t)bidx * A_ * A_;
#pragma unroll
  for (int r = 0; r < 8; ++r) {
    int m = mb + r;
    drow[(size_t)m * A_ + n] = sqb[m] + sqb[n] - 2.f * acc[r];
  }
}

// ---------------- stable top-8 smallest + sorted indices + group row ----------------
__global__ void topk_group(const float* __restrict__ dists, const float* __restrict__ isinit,
                           int* __restrict__ idx, float* __restrict__ group) {
  int row = blockIdx.x * blockDim.x + threadIdx.x;
  if (row >= ROWS_) return;
  const float* d = dists + (size_t)row * A_;
  unsigned chosen[4] = {0u, 0u, 0u, 0u};
  int sel[MG_];
  for (int s = 0; s < MG_; ++s) {
    float best = 3.4e38f; int bi = 0;
    for (int c = 0; c < A_; ++c) {
      if (chosen[c >> 5] & (1u << (c & 31))) continue;
      float v = d[c];
      if (v < best) { best = v; bi = c; }   // ascending scan + strict < == stable argsort
    }
    chosen[bi >> 5] |= 1u << (bi & 31);
    sel[s] = bi;
  }
  for (int i = 1; i < MG_; ++i) {
    int key = sel[i], j = i - 1;
    while (j >= 0 && sel[j] > key) { sel[j + 1] = sel[j]; --j; }
    sel[j + 1] = key;
  }
  for (int s = 0; s < MG_; ++s) idx[row * MG_ + s] = sel[s];
  float gi = isinit[row];
  float* grow = group + (size_t)row * A_;
  for (int c = 0; c < A_; ++c) grow[c] = 0.f;
  for (int s = 0; s < MG_; ++s) grow[sel[s]] = gi;
}

// ---------------- LSTM: input projection GEMM with gather (bf16) ----------------
__global__ void lstm_input_gemm(const bf16_t* __restrict__ newTb, const int* __restrict__ idx,
                                int agent,
                                const bf16_t* __restrict__ wih_f, const bf16_t* __restrict__ wih_r,
                                const float* __restrict__ bih_f, const float* __restrict__ bhh_f,
                                const float* __restrict__ bih_r, const float* __restrict__ bhh_r,
                                float* __restrict__ Zx) {
  int lane   = threadIdx.x & 31;
  int waveId = blockIdx.x * 4 + (threadIdx.x >> 5);   // tiles: 32 x 64
  int tm = waveId / 64, tn = waveId % 64;
  int m = tm * 16 + (lane & 15);
  int bb = m >> 3, j = m & 7;
  int src = bb * A_ + idx[(bb * A_ + agent) * MG_ + j];
  const bf16_t* pa = newTb + (size_t)src * TD_;
  int n16 = tn * 16 + (lane & 15);
  int dir = n16 >> 9;
  int gn  = n16 & (GATES_ - 1);
  const bf16_t* pw = (dir ? wih_r : wih_f) + (size_t)gn * TD_;
  v8f acc = {};
  for (int k = 0; k < TD_; k += 32) {
    acc = wmma32(load_a_frag(pa, k, lane), load_b_frag(pw, k, lane), acc);
  }
  float bv = dir ? (bih_r[gn] + bhh_r[gn]) : (bih_f[gn] + bhh_f[gn]);
  int mb = tm * 16 + ((lane >> 4) << 3);
#pragma unroll
  for (int r = 0; r < 8; ++r)
    Zx[(size_t)(mb + r) * (2 * GATES_) + n16] = acc[r] + bv;
}

// ---------------- LSTM recurrence: one 32-wave workgroup per direction ----------------
// whh B-fragments hoisted into registers once (reused across all 4 M tiles x 8 timesteps);
// h stays in LDS as bf16 between timesteps.
__global__ void lstm_recur(const float* __restrict__ Zx,
                           const bf16_t* __restrict__ whh_f, const bf16_t* __restrict__ whh_r,
                           float* __restrict__ hs) {   // (2, 8, 64, 128)
  int dir = blockIdx.x;
  const bf16_t* whh = dir ? whh_r : whh_f;
  extern __shared__ char smem[];
  float*  zbuf = (float*)smem;                                   // 131072 B
  float*  cbuf = (float*)(smem + 131072);                        //  32768 B
  bf16_t* hbuf = (bf16_t*)(smem + 131072 + 32768);               //  16384 B
  int tid  = threadIdx.x;           // 1024
  int lane = tid & 31;
  int tn   = tid >> 5;              // wave id == N tile (32 tiles of 16)

  v16bf bfrag[4];                   // this wave's 16-gate column slice of whh (K=128)
  {
    const bf16_t* pw = whh + (size_t)(tn * 16 + (lane & 15)) * H_;
#pragma unroll
    for (int kk = 0; kk < 4; ++kk) bfrag[kk] = load_b_frag(pw, kk * 32, lane);
  }

  for (int p = tid; p < B_ * H_; p += 1024) { cbuf[p] = 0.f; hbuf[p] = (bf16_t)0.f; }
  __syncthreads();
  for (int t = 0; t < MG_; ++t) {
    int tt = dir ? (MG_ - 1 - t) : t;
    for (int tm = 0; tm < 4; ++tm) {          // z = h @ whh^T  (64 x 512, K=128)
      const bf16_t* ph = hbuf + (size_t)(tm * 16 + (lane & 15)) * H_;
      v8f acc = {};
#pragma unroll
      for (int kk = 0; kk < 4; ++kk)
        acc = wmma32(load_a_frag(ph, kk * 32, lane), bfrag[kk], acc);
      int n  = tn * 16 + (lane & 15);
      int mb = tm * 16 + ((lane >> 4) << 3);
#pragma unroll
      for (int r = 0; r < 8; ++r) {
        int bb = mb + r;
        zbuf[bb * GATES_ + n] =
            acc[r] + Zx[(size_t)(bb * MG_ + tt) * (2 * GATES_) + dir * GATES_ + n];
      }
    }
    __syncthreads();
    for (int p = tid; p < B_ * H_; p += 1024) {   // gates i,f,g,o
      int bb = p >> 7, hh = p & (H_ - 1);
      const float* z = zbuf + bb * GATES_;
      float zi = z[hh], zf = z[H_ + hh], zg = z[2 * H_ + hh], zo = z[3 * H_ + hh];
      float c = sigm(zf) * cbuf[p] + sigm(zi) * tanhf(zg);
      float hv = sigm(zo) * tanhf(c);
      cbuf[p] = c;
      hbuf[p] = (bf16_t)hv;
      hs[(size_t)((dir * MG_ + tt) * B_ + bb) * H_ + hh] = hv;
    }
    __syncthreads();
  }
}

// ---------------- masked scatter of bi-LSTM outputs (f32 + bf16 mirrors) ----------------
__global__ void lstm_scatter(float* __restrict__ newT, bf16_t* __restrict__ newTb,
                             const int* __restrict__ idx, const float* __restrict__ isinit,
                             const float* __restrict__ hs, int agent) {
  int t = blockIdx.x * blockDim.x + threadIdx.x;     // 64*8*256
  if (t >= B_ * MG_ * TD_) return;
  int col = t & (TD_ - 1);
  int j   = (t >> 8) & (MG_ - 1);
  int bb  = t >> 11;
  if (isinit[bb * A_ + agent] == 0.f) return;
  int dst = bb * A_ + idx[(bb * A_ + agent) * MG_ + j];
  float v;
  if (col < H_) v = hs[(size_t)((0 * MG_ + j) * B_ + bb) * H_ + col];
  else          v = hs[(size_t)((1 * MG_ + j) * B_ + bb) * H_ + (col - H_)];
  newT [(size_t)dst * TD_ + col] = v;
  newTb[(size_t)dst * TD_ + col] = (bf16_t)v;
}

// ---------------- actor_2 input: relu(concat(ct, newT)) -> bf16 ----------------
__global__ void concat_relu(const float* __restrict__ ct, const float* __restrict__ nt,
                            bf16_t* __restrict__ Z) {
  int t = blockIdx.x * blockDim.x + threadIdx.x;     // 8192*512
  if (t >= ROWS_ * 2 * TD_) return;
  int col = t & (2 * TD_ - 1);
  int r   = t >> 9;
  float v = (col < TD_) ? ct[(size_t)r * TD_ + col] : nt[(size_t)r * TD_ + (col - TD_)];
  Z[t] = (bf16_t)fmaxf(v, 0.f);
}

// ---------------- host-side orchestration ----------------
extern "C" void kernel_launch(void* const* d_in, const int* in_sizes, int n_in,
                              void* d_out, int out_size, void* d_ws, size_t ws_size,
                              hipStream_t stream) {
  (void)in_sizes; (void)n_in; (void)out_size; (void)ws_size;
  const float* obs    = (const float*)d_in[0];
  const float* a1_w1  = (const float*)d_in[1];
  const float* a1_b1  = (const float*)d_in[2];
  const float* a1_g1  = (const float*)d_in[3];
  const float* a1_be1 = (const float*)d_in[4];
  const float* a1_w2  = (const float*)d_in[5];
  const float* a1_b2  = (const float*)d_in[6];
  const float* a1_g2  = (const float*)d_in[7];
  const float* a1_be2 = (const float*)d_in[8];
  const float* a2_w1  = (const float*)d_in[9];
  const float* a2_b1  = (const float*)d_in[10];
  const float* a2_g1  = (const float*)d_in[11];
  const float* a2_be1 = (const float*)d_in[12];
  const float* a2_w2  = (const float*)d_in[13];
  const float* a2_b2  = (const float*)d_in[14];
  const float* a2_g2  = (const float*)d_in[15];
  const float* a2_be2 = (const float*)d_in[16];
  const float* at_w1  = (const float*)d_in[17];
  const float* at_b1  = (const float*)d_in[18];
  const float* at_w2  = (const float*)d_in[19];
  const float* at_b2  = (const float*)d_in[20];
  const float* at_w3  = (const float*)d_in[21];
  const float* at_b3  = (const float*)d_in[22];
  const float* wih_f  = (const float*)d_in[23];
  const float* whh_f  = (const float*)d_in[24];
  const float* bih_f  = (const float*)d_in[25];
  const float* bhh_f  = (const float*)d_in[26];
  const float* wih_r  = (const float*)d_in[27];
  const float* whh_r  = (const float*)d_in[28];
  const float* bih_r  = (const float*)d_in[29];
  const float* bhh_r  = (const float*)d_in[30];

  float* out    = (float*)d_out;
  float* action = out;                         // (B,A,64)
  float* group  = out + 524288;                // (B,A,A)
  float* initp  = out + 1572864;               // (B,A)
  float* isinit = out + 1581056;               // (B,A)
  float* newT   = out + 1589248;               // (B,A,256)
  float* oldT   = out + 3686400;               // (B,A,256) == ct

  char* ws = (char*)d_ws;
  bf16_t* OBSB  = (bf16_t*)ws;                        // 16 MB (phase 1)
  bf16_t* Z512B = (bf16_t*)ws;                        //  8 MB (actor_2, reuse)
  bf16_t* T1B   = (bf16_t*)(ws + 8388608);            //  4 MB (actor_2, reuse)
  float*  H1F   = (float*) (ws + 16777216);           //  8 MB
  float*  AT2   = (float*) (ws + 16777216);           //  4 MB (reuse)
  float*  DIST  = (float*) (ws + 20971520);           //  4 MB (reuse)
  bf16_t* H1B   = (bf16_t*)(ws + 25165824);           //  4 MB
  float*  T2    = (float*) (ws + 25165824);           //  2 MB (reuse)
  float*  CTRAW = (float*) (ws + 29360128);           //  8 MB
  float*  T1    = (float*) (ws + 29360128);           //  8 MB (reuse)
  bf16_t* CTB   = (bf16_t*)(ws + 37748736);           //  4 MB
  bf16_t* NEWTB = (bf16_t*)(ws + 41943040);           //  4 MB
  bf16_t* AT1B  = (bf16_t*)(ws + 46137344);           //  2 MB
  float*  SQ    = (float*) (ws + 48234496);           // 32 KB
  int*    IDX   = (int*)   (ws + 48267264);           // 256 KB
  float*  ZX    = (float*) (ws + 48529408);           //  2 MB
  float*  HS    = (float*) (ws + 50626560);           // 512 KB
  bf16_t* Wa1w1b = (bf16_t*)(ws + 51150848);
  bf16_t* Wa1w2b = (bf16_t*)(ws + 51675136);
  bf16_t* Watw1b = (bf16_t*)(ws + 51806208);
  bf16_t* Watw2b = (bf16_t*)(ws + 51871744);
  bf16_t* Wa2w1b = (bf16_t*)(ws + 51904512);
  bf16_t* Wa2w2b = (bf16_t*)(ws + 52166656);
  bf16_t* Wwihfb = (bf16_t*)(ws + 52199424);
  bf16_t* Wwihrb = (bf16_t*)(ws + 52461568);
  bf16_t* Wwhhfb = (bf16_t*)(ws + 52723712);
  bf16_t* Wwhhrb = (bf16_t*)(ws + 52854784);

  // ---- one-time bf16 conversions ----
  cvt_bf16<<<32768, 256, 0, stream>>>(obs, OBSB, ROWS_ * OBS_);
  cvt_bf16<<<1024, 256, 0, stream>>>(a1_w1, Wa1w1b, TD_ * OBS_);
  cvt_bf16<<<256,  256, 0, stream>>>(a1_w2, Wa1w2b, TD_ * TD_);
  cvt_bf16<<<128,  256, 0, stream>>>(at_w1, Watw1b, EATT_ * TD_);
  cvt_bf16<<<64,   256, 0, stream>>>(at_w2, Watw2b, EATT_ * EATT_);
  cvt_bf16<<<512,  256, 0, stream>>>(a2_w1, Wa2w1b, TD_ * 2 * TD_);
  cvt_bf16<<<64,   256, 0, stream>>>(a2_w2, Wa2w2b, ACTD_ * TD_);
  cvt_bf16<<<512,  256, 0, stream>>>(wih_f, Wwihfb, GATES_ * TD_);
  cvt_bf16<<<512,  256, 0, stream>>>(wih_r, Wwihrb, GATES_ * TD_);
  cvt_bf16<<<256,  256, 0, stream>>>(whh_f, Wwhhfb, GATES_ * H_);
  cvt_bf16<<<256,  256, 0, stream>>>(whh_r, Wwhhrb, GATES_ * H_);

  // ---- actor_1 ----
  gemm_coop<0><<<2048, 128, 2 * BUFB_, stream>>>(OBSB, OBS_, Wa1w1b, OBS_, a1_b1,
                                                 H1F, (bf16_t*)nullptr, TD_, ROWS_, TD_, OBS_);
  ln_kernel<TD_, 1><<<2048, 128, 0, stream>>>(H1F, a1_g1, a1_be1, (float*)nullptr, H1B, ROWS_);
  gemm_coop<0><<<2048, 128, 2 * BUFB_, stream>>>(H1B, TD_, Wa1w2b, TD_, a1_b2,
                                                 CTRAW, (bf16_t*)nullptr, TD_, ROWS_, TD_, TD_);
  ln_kernel<TD_, 0><<<2048, 128, 0, stream>>>(CTRAW, a1_g2, a1_be2, oldT, CTB, ROWS_);
  hipMemcpyAsync(newT, oldT, (size_t)ROWS_ * TD_ * sizeof(float),
                 hipMemcpyDeviceToDevice, stream);
  hipMemcpyAsync(NEWTB, CTB, (size_t)ROWS_ * TD_ * sizeof(bf16_t),
                 hipMemcpyDeviceToDevice, stream);

  // ---- attention unit ----
  gemm_coop<1><<<1024, 128, 2 * BUFB_, stream>>>(CTB, TD_, Watw1b, TD_, at_b1,
                                                 (float*)nullptr, AT1B, EATT_, ROWS_, EATT_, TD_);
  gemm_coop<1><<<1024, 128, 2 * BUFB_, stream>>>(AT1B, EATT_, Watw2b, EATT_, at_b2,
                                                 AT2, (bf16_t*)nullptr, EATT_, ROWS_, EATT_, EATT_);
  attn_final<<<32, 256, 0, stream>>>(AT2, at_w3, at_b3, initp, isinit);

  // ---- pairwise distances + top-8 groups ----
  rownorm_kernel<<<32, 256, 0, stream>>>(oldT, SQ);
  dist_gemm<<<dim3(16, B_), 128, 0, stream>>>(CTB, SQ, DIST);
  topk_group<<<64, 128, 0, stream>>>(DIST, isinit, IDX, group);

  // ---- sequential per-agent communication scan (128 steps) ----
  for (int agent = 0; agent < A_; ++agent) {
    lstm_input_gemm<<<512, 128, 0, stream>>>(NEWTB, IDX, agent, Wwihfb, Wwihrb,
                                             bih_f, bhh_f, bih_r, bhh_r, ZX);
    lstm_recur<<<2, 1024, 180224, stream>>>(ZX, Wwhhfb, Wwhhrb, HS);
    lstm_scatter<<<512, 256, 0, stream>>>(newT, NEWTB, IDX, isinit, HS, agent);
  }

  // ---- actor_2 ----
  concat_relu<<<16384, 256, 0, stream>>>(oldT, newT, Z512B);
  gemm_coop<0><<<2048, 128, 2 * BUFB_, stream>>>(Z512B, 2 * TD_, Wa2w1b, 2 * TD_, a2_b1,
                                                 T1, (bf16_t*)nullptr, TD_, ROWS_, TD_, 2 * TD_);
  ln_kernel<TD_, 0><<<2048, 128, 0, stream>>>(T1, a2_g1, a2_be1, (float*)nullptr, T1B, ROWS_);
  gemm_coop<0><<<512, 128, 2 * BUFB_, stream>>>(T1B, TD_, Wa2w2b, TD_, a2_b2,
                                                T2, (bf16_t*)nullptr, ACTD_, ROWS_, ACTD_, TD_);
  ln_kernel<ACTD_, 2><<<2048, 128, 0, stream>>>(T2, a2_g2, a2_be2, action, (bf16_t*)nullptr, ROWS_);
}
